// Encoder_34540126994448
// MI455X (gfx1250) — matrix operand order repaired and checked by reading, not compiled, not run on hardware
//
#include <hip/hip_runtime.h>
#include <hip/hip_bf16.h>

// ---------------------------------------------------------------------------
// Types for CDNA5 WMMA (wave32): v_wmma_f32_16x16x32_bf16
// ---------------------------------------------------------------------------
typedef __attribute__((ext_vector_type(16))) __bf16 v16bf;
typedef __attribute__((ext_vector_type(8)))  __bf16 v8bf;
typedef __attribute__((ext_vector_type(8)))  float  v8f;

static __host__ __device__ inline int cdiv(int a, int b) { return (a + b - 1) / b; }

// ---------------------------------------------------------------------------
// Prep kernels
// ---------------------------------------------------------------------------
__global__ void k_f32_to_bf16(const float* __restrict__ in, __bf16* __restrict__ out, int n) {
    int i = blockIdx.x * blockDim.x + threadIdx.x;
    if (i < n) out[i] = (__bf16)in[i];
}

// W[k][n] (row-major KxN) -> Wt[n][k] bf16: makes the WMMA B-fragment a
// contiguous 32B read per lane instead of 16 strided u16 loads.
__global__ void k_w_to_bf16_t(const float* __restrict__ W, __bf16* __restrict__ Wt,
                              int K, int cols) {
    int i = blockIdx.x * blockDim.x + threadIdx.x;
    if (i >= K * cols) return;
    int k = i / cols;
    int n = i - k * cols;
    Wt[n * K + k] = (__bf16)W[i];
}

__global__ void k_fill1(float* __restrict__ p, int n) {
    int i = blockIdx.x * blockDim.x + threadIdx.x;
    if (i < n) p[i] = 1.0f;            // self-loop contributes 1 to degree
}

__global__ void k_deg_count(const int* __restrict__ dst, float* __restrict__ deg, int e) {
    int i = blockIdx.x * blockDim.x + threadIdx.x;
    if (i < e) atomicAdd(&deg[dst[i]], 1.0f);
}

__global__ void k_rsqrt_inplace(float* __restrict__ p, int n) {
    int i = blockIdx.x * blockDim.x + threadIdx.x;
    if (i < n) p[i] = rsqrtf(p[i]);
}

// agg[i, f..f+3] = h * dinv[node]^2   (self-loop term; initializes agg, no memset needed)
__global__ void k_self_init4(const float4* __restrict__ h4, const float* __restrict__ dinv,
                             float4* __restrict__ agg4, int cols4, int total4) {
    int i = blockIdx.x * blockDim.x + threadIdx.x;
    if (i >= total4) return;
    float d = dinv[i / cols4];
    float d2 = d * d;
    float4 v = h4[i];
    v.x *= d2; v.y *= d2; v.z *= d2; v.w *= d2;
    agg4[i] = v;
}

// edge-parallel scatter, 4 features per thread: amortizes src/dst/dinv loads
// and address math over 4 global_atomic_add_f32. Random traffic is L2-resident
// (feature matrix 25.6MB << 192MB global L2).
__global__ void k_scatter4(const float4* __restrict__ h4, const float* __restrict__ dinv,
                           const int* __restrict__ src, const int* __restrict__ dst,
                           float* __restrict__ agg, int cols4, int total4) {
    int i = blockIdx.x * blockDim.x + threadIdx.x;
    if (i >= total4) return;
    int e  = i / cols4;
    int f4 = i - e * cols4;
    int s = src[e];
    int d = dst[e];
    float nrm = dinv[s] * dinv[d];
    float4 v = h4[s * cols4 + f4];
    float* base = agg + (d * cols4 + f4) * 4;
    atomicAdd(base + 0, v.x * nrm);
    atomicAdd(base + 1, v.y * nrm);
    atomicAdd(base + 2, v.z * nrm);
    atomicAdd(base + 3, v.w * nrm);
}

// out_bf16[i, f..f+3] = bf16(relu(agg + b))   (feeds next layer's WMMA A operand)
__global__ void k_bias_relu_bf16_4(const float4* __restrict__ agg4, const float* __restrict__ bias,
                                   __bf16* __restrict__ out, int cols4, int total4) {
    int i = blockIdx.x * blockDim.x + threadIdx.x;
    if (i >= total4) return;
    int f4 = (i % cols4) * 4;
    float4 v = agg4[i];
    const float* b = bias + f4;
    __bf16* o = out + i * 4;
    o[0] = (__bf16)fmaxf(v.x + b[0], 0.0f);
    o[1] = (__bf16)fmaxf(v.y + b[1], 0.0f);
    o[2] = (__bf16)fmaxf(v.z + b[2], 0.0f);
    o[3] = (__bf16)fmaxf(v.w + b[3], 0.0f);
}

// out[i, f..f+3] = relu(out + b)  in place (final layer)
__global__ void k_bias_relu_inplace4(float4* __restrict__ out4, const float* __restrict__ bias,
                                     int cols4, int total4) {
    int i = blockIdx.x * blockDim.x + threadIdx.x;
    if (i >= total4) return;
    int f4 = (i % cols4) * 4;
    float4 v = out4[i];
    const float* b = bias + f4;
    v.x = fmaxf(v.x + b[0], 0.0f);
    v.y = fmaxf(v.y + b[1], 0.0f);
    v.z = fmaxf(v.z + b[2], 0.0f);
    v.w = fmaxf(v.w + b[3], 0.0f);
    out4[i] = v;
}

// ---------------------------------------------------------------------------
// WMMA GEMM: C[rows, cols] = A[rows, 128] @ W[128, cols],  W given transposed
// as Wt[cols, 128]. A bf16 row-major, C f32. One wave computes one 16x16 tile;
// K=128 -> 4 x v_wmma_f32_16x16x32_bf16 per tile.
// blockDim.x = 2*cols (wave w -> column tile w), gridDim.x = rows/16
// (rows = 50000 = 3125*16 exactly).
//
// A tile (16x128 bf16 = 4KB) is shared by every wave in the block -> staged
// once in LDS cooperatively, fragments read back with ds_load_b128.
//
// Fragment layouts used here (wave32; A and B share contiguous K order):
//  A 16x32 bf16 : lane m in [0,15]  -> M=m,  slots 0..15 = K 0..15
//                 lane 16+m         -> M=m,  slots 0..15 = K 16..31
//  B 32x16 bf16 : lane -> column (lane&15); lane half selects K half; slot s = K(half*16+s)
//                 -> with Wt this is one contiguous 32B read per lane.
//  C/D 16x16 f32: lane -> column; VGPR r -> row r + 8*(lane>>4)
// ---------------------------------------------------------------------------
__global__ void gcn_wmma_gemm(const __bf16* __restrict__ A, const __bf16* __restrict__ Wt,
                              float* __restrict__ C, int cols) {
    const int K = 128;
    __shared__ __bf16 As[16 * 128];    // 4KB of the 320KB/WGP LDS

    // cooperative stage of the A tile: 256 x uint4 (16B) covers 16*128 bf16
    {
        const uint4* gsrc = reinterpret_cast<const uint4*>(A + (size_t)blockIdx.x * 16 * K);
        uint4* lds = reinterpret_cast<uint4*>(As);
        for (int i = threadIdx.x; i < (16 * K) / 8; i += blockDim.x)
            lds[i] = gsrc[i];
    }
    __syncthreads();

    const int wave = threadIdx.x >> 5;
    const int lane = threadIdx.x & 31;
    const int hi   = lane >> 4;        // lane half (0 or 1)
    const int m    = lane & 15;

    const int n0 = wave * 16;          // column tile base

    const __bf16* ap = As + m * K;               // this lane's A row in LDS
    const __bf16* wp = Wt + (n0 + m) * K;        // this lane's B column (contiguous K)

    v8f acc = {};
#pragma unroll
    for (int k0 = 0; k0 < K; k0 += 32) {
        // --- A fragment: two 16B ds_load_b128 ---
        const int kbA = k0 + hi * 16;
        v8bf alo = *reinterpret_cast<const v8bf*>(ap + kbA);        // K = kbA..kbA+7
        v8bf ahi = *reinterpret_cast<const v8bf*>(ap + kbA + 8);    // K = kbA+8..kbA+15
        v16bf a;
#pragma unroll
        for (int e = 0; e < 8; ++e) { a[e] = alo[e]; a[8 + e] = ahi[e]; }

        // --- B fragment: one contiguous 32B read (2 x global_load_b128) ---
        const int kbB = k0 + hi * 16;
        v16bf b = *reinterpret_cast<const v16bf*>(wp + kbB);

        acc = __builtin_amdgcn_wmma_f32_16x16x32_bf16(
            /*neg_a=*/false, a, /*neg_b=*/false, b,
            /*c_mod=*/(short)0, acc, /*reuse_a=*/false, /*reuse_b=*/false);
    }

    float* cp = C + (blockIdx.x * 16 + hi * 8) * cols + n0 + m;
#pragma unroll
    for (int r = 0; r < 8; ++r) cp[r * cols] = acc[r];
}

// ---------------------------------------------------------------------------
// Launcher
// ---------------------------------------------------------------------------
extern "C" void kernel_launch(void* const* d_in, const int* in_sizes, int n_in,
                              void* d_out, int out_size, void* d_ws, size_t ws_size,
                              hipStream_t stream) {
    const float* x  = (const float*)d_in[0];
    const int*   ei = (const int*)  d_in[1];
    const float* W1 = (const float*)d_in[2];
    const float* b1 = (const float*)d_in[3];
    const float* W2 = (const float*)d_in[4];
    const float* b2 = (const float*)d_in[5];
    float* out = (float*)d_out;

    const int IN = 128, HID = 128, OUT = 64;
    const int N = in_sizes[0] / IN;
    const int E = in_sizes[1] / 2;
    const int* src = ei;        // edge_index[0]
    const int* dst = ei + E;    // edge_index[1]

    // workspace carve-out (256B aligned)
    char* p = (char*)d_ws;
    auto carve = [&](size_t bytes) {
        char* r = p;
        p += (bytes + 255) & ~(size_t)255;
        return r;
    };
    __bf16* Xh   = (__bf16*)carve((size_t)N * IN * sizeof(__bf16));  // reused as bf16(h1)
    __bf16* W1t  = (__bf16*)carve((size_t)IN * HID * sizeof(__bf16));   // transposed
    __bf16* W2t  = (__bf16*)carve((size_t)HID * OUT * sizeof(__bf16));  // transposed
    float*  dinv = (float*) carve((size_t)N * sizeof(float));
    float*  h1   = (float*) carve((size_t)N * HID * sizeof(float));
    float*  agg1 = (float*) carve((size_t)N * HID * sizeof(float));
    float*  h2   = (float*) carve((size_t)N * OUT * sizeof(float));

    const int T = 256;
    int tot;

    // 0) precision prep (+ W transpose for the WMMA B operand)
    tot = N * IN;    k_f32_to_bf16<<<cdiv(tot, T), T, 0, stream>>>(x, Xh, tot);
    tot = IN * HID;  k_w_to_bf16_t<<<cdiv(tot, T), T, 0, stream>>>(W1, W1t, IN, HID);
    tot = HID * OUT; k_w_to_bf16_t<<<cdiv(tot, T), T, 0, stream>>>(W2, W2t, HID, OUT);

    // 1) degrees (incl. self-loop) -> dinv = deg^{-1/2}
    k_fill1        <<<cdiv(N, T), T, 0, stream>>>(dinv, N);
    k_deg_count    <<<cdiv(E, T), T, 0, stream>>>(dst, dinv, E);
    k_rsqrt_inplace<<<cdiv(N, T), T, 0, stream>>>(dinv, N);

    // 2) layer 1: h1 = X @ W1   (8 waves/block -> 128 cols)
    gcn_wmma_gemm<<<dim3(N / 16), dim3(2 * HID), 0, stream>>>(Xh, W1t, h1, HID);

    // 3) layer 1 aggregation: agg1 = D^{-1/2} A D^{-1/2} h1 + D^{-1} h1
    {
        const int c4 = HID / 4;
        tot = N * c4;
        k_self_init4<<<cdiv(tot, T), T, 0, stream>>>((const float4*)h1, dinv,
                                                     (float4*)agg1, c4, tot);
        tot = E * c4;
        k_scatter4<<<cdiv(tot, T), T, 0, stream>>>((const float4*)h1, dinv, src, dst,
                                                   agg1, c4, tot);
        tot = N * c4;
        k_bias_relu_bf16_4<<<cdiv(tot, T), T, 0, stream>>>((const float4*)agg1, b1,
                                                           Xh /*now bf16(relu h1)*/, c4, tot);
    }

    // 4) layer 2: h2 = relu(h1) @ W2   (4 waves/block -> 64 cols)
    gcn_wmma_gemm<<<dim3(N / 16), dim3(2 * OUT), 0, stream>>>(Xh, W2t, h2, OUT);

    // 5) layer 2 aggregation directly into d_out
    {
        const int c4 = OUT / 4;
        tot = N * c4;
        k_self_init4<<<cdiv(tot, T), T, 0, stream>>>((const float4*)h2, dinv,
                                                     (float4*)out, c4, tot);
        tot = E * c4;
        k_scatter4<<<cdiv(tot, T), T, 0, stream>>>((const float4*)h2, dinv, src, dst,
                                                   out, c4, tot);
        tot = N * c4;
        k_bias_relu_inplace4<<<cdiv(tot, T), T, 0, stream>>>((float4*)out, b2, c4, tot);
    }
}
